// GCNModel_28681791603240
// MI455X (gfx1250) — matrix operand order repaired and checked by reading, not compiled, not run on hardware
//
#include <hip/hip_runtime.h>

#define DFEAT 128
#define WPAD  136   // LDS row stride in halves (padded: 4-bank spacing per row)

typedef __attribute__((ext_vector_type(8)))  __bf16 v8bf;
typedef __attribute__((ext_vector_type(16))) __bf16 v16bf;
typedef __attribute__((ext_vector_type(8)))  float  v8f;

__device__ __forceinline__ __bf16 f2bf(float f) {
  unsigned int u = __float_as_uint(f);
  unsigned short h = (unsigned short)((u + 0x7fffu + ((u >> 16) & 1u)) >> 16); // RNE
  return __builtin_bit_cast(__bf16, h);
}

// concat two 8-half LDS chunks into one 16-half fragment register set
__device__ __forceinline__ v16bf ld16(const __bf16* p0, const __bf16* p1) {
  v8bf lo = *(const v8bf*)p0;
  v8bf hi = *(const v8bf*)p1;
  return __builtin_shufflevector(lo, hi, 0,1,2,3,4,5,6,7,8,9,10,11,12,13,14,15);
}

// ---------------- small utility kernels ----------------
__global__ void zero_i32_kernel(int* p, int n) {
  int i = blockIdx.x * blockDim.x + threadIdx.x;
  if (i < n) p[i] = 0;
}
__global__ void zero_f32_kernel(float* p, int n) {
  int i = blockIdx.x * blockDim.x + threadIdx.x;
  if (i < n) p[i] = 0.0f;
}

// in-degree histogram over dst (self-loops accounted for later as +1)
__global__ void count_edges_kernel(const int* __restrict__ dst, int* __restrict__ deg, int E) {
  int e = blockIdx.x * blockDim.x + threadIdx.x;
  if (e < E) atomicAdd(&deg[dst[e]], 1);
}

// single-block exclusive scan of deg -> offs (and cursor copy)
__global__ __launch_bounds__(1024) void scan_offsets_kernel(const int* __restrict__ deg, int n,
                                                            int* __restrict__ offs,
                                                            int* __restrict__ cursor) {
  __shared__ int sdata[1024];
  __shared__ int s_carry;
  const int tid = threadIdx.x;
  if (tid == 0) s_carry = 0;
  __syncthreads();
  for (int base = 0; base < n; base += 1024) {
    int i = base + tid;
    int v = (i < n) ? deg[i] : 0;
    sdata[tid] = v;
    __syncthreads();
    for (int off = 1; off < 1024; off <<= 1) {
      int t = (tid >= off) ? sdata[tid - off] : 0;
      __syncthreads();
      sdata[tid] += t;
      __syncthreads();
    }
    int carry = s_carry;
    int excl = carry + sdata[tid] - v;
    if (i < n) { offs[i] = excl; cursor[i] = excl; }
    __syncthreads();
    if (tid == 0) s_carry = carry + sdata[1023];
    __syncthreads();
  }
  if (tid == 0) offs[n] = s_carry;
}

// dinv[i] = rsqrt(in_degree + 1 self loop)
__global__ void dinv_kernel(const int* __restrict__ deg, float* __restrict__ dinv, int n) {
  int i = blockIdx.x * blockDim.x + threadIdx.x;
  if (i < n) dinv[i] = rsqrtf((float)(deg[i] + 1));
}

// scatter edges into CSR-by-destination
__global__ void scatter_edges_kernel(const int* __restrict__ src, const int* __restrict__ dst,
                                     int* __restrict__ cursor, int* __restrict__ csrSrc, int E) {
  int e = blockIdx.x * blockDim.x + threadIdx.x;
  if (e < E) {
    int d = dst[e];
    int pos = atomicAdd(&cursor[d], 1);
    csrSrc[pos] = src[e];
  }
}

// ---------------- WMMA GEMM: Y[N x 128] = X[N x 128] @ W[128 x 128] ----------------
// 256 threads = 8 waves; wave w owns column tile w (16 cols). W is transposed +
// bf16-converted into LDS once per block; B fragments hoisted into registers.
__global__ __launch_bounds__(256)
void gemm_xw_wmma_kernel(const float* __restrict__ X, const float* __restrict__ W,
                         float* __restrict__ Y, int nRows) {
  __shared__ __align__(16) __bf16 Wt[DFEAT * WPAD];  // Wt[n][k]
  __shared__ __align__(16) __bf16 Xb[16 * WPAD];     // Xb[r][k]
  const int tid = threadIdx.x;

  // stage W transposed as bf16: thread owns column n, 8 consecutive k per chunk.
  // Global reads coalesced over n; one 16B ds_store_b128 per chunk.
  {
    const int n = tid & 127;
    for (int k0 = (tid >> 7) * 8; k0 < DFEAT; k0 += 16) {
      v8bf w;
#pragma unroll
      for (int j = 0; j < 8; ++j) w[j] = f2bf(W[(k0 + j) * DFEAT + n]);
      *(v8bf*)&Wt[n * WPAD + k0] = w;
    }
  }
  __syncthreads();

  const int lane = tid & 31;
  const int wv   = tid >> 5;    // column tile 0..7
  const int lm   = lane & 15;
  const int half = lane >> 4;

  // B fragments (32x16 bf16): lane<16 holds K=k0..k0+15 of col lm; lane>=16 K=k0+16..k0+31
  v16bf Bf[4];
#pragma unroll
  for (int s = 0; s < 4; ++s) {
    const __bf16* p = &Wt[(wv * 16 + lm) * WPAD + s * 32 + half * 16];
    Bf[s] = ld16(p, p + 8);
  }

  // X staging map: thread owns row r = tid>>4, 8 floats at k0 = (tid&15)*8
  const int sr  = tid >> 4;
  const int sk0 = (tid & 15) * 8;
  __bf16* xdst = &Xb[sr * WPAD + sk0];

  const int nTiles = (nRows + 15) >> 4;
  for (int t = blockIdx.x; t < nTiles; t += gridDim.x) {
    __syncthreads();  // protect Xb from previous iteration's readers
    {
      const int row = t * 16 + sr;
      v8bf xv;
      if (row < nRows) {
        const float4* xp = (const float4*)(X + (size_t)row * DFEAT + sk0);
        float4 f0 = xp[0], f1 = xp[1];
        xv[0] = f2bf(f0.x); xv[1] = f2bf(f0.y); xv[2] = f2bf(f0.z); xv[3] = f2bf(f0.w);
        xv[4] = f2bf(f1.x); xv[5] = f2bf(f1.y); xv[6] = f2bf(f1.z); xv[7] = f2bf(f1.w);
      } else {
        xv = (v8bf)(__bf16)0.0f;
      }
      *(v8bf*)xdst = xv;
    }
    __syncthreads();

    // prefetch all A fragments (8x ds_load_b128), then the WMMA accumulation chain
    v16bf a[4];
#pragma unroll
    for (int s = 0; s < 4; ++s) {
      const __bf16* pa = &Xb[lm * WPAD + s * 32 + half * 16];
      a[s] = ld16(pa, pa + 8);
    }
    v8f c = {};
#pragma unroll
    for (int s = 0; s < 4; ++s)
      c = __builtin_amdgcn_wmma_f32_16x16x32_bf16(false, a[s], false, Bf[s],
                                                  (short)0, c, false, false);

    // C/D layout: VGPR r -> M = r + 8*half, N = lane&15
    const int rowBase = t * 16 + half * 8;
    float* Yp = Y + (size_t)rowBase * DFEAT + wv * 16 + lm;
    if (rowBase + 8 <= nRows) {
#pragma unroll
      for (int r = 0; r < 8; ++r) Yp[(size_t)r * DFEAT] = c[r];
    } else {
#pragma unroll
      for (int r = 0; r < 8; ++r)
        if (rowBase + r < nRows) Yp[(size_t)r * DFEAT] = c[r];
    }
  }
}

// ---------------- gather-only aggregation + bias + ReLU ----------------
// out[d] = relu( dinv[d] * ( sum_{s in N(d)} dinv[s]*H[s] + dinv[d]*H[d] ) + b )
// One wave per destination node; 4 floats per lane (128 features).
__global__ __launch_bounds__(256)
void gcn_aggregate_kernel(const float* __restrict__ H, const int* __restrict__ offs,
                          const int* __restrict__ csrSrc, const float* __restrict__ dinv,
                          const float* __restrict__ bias, float* __restrict__ Y, int nNodes) {
  int wv = (blockIdx.x * 256 + threadIdx.x) >> 5;
  int lane = threadIdx.x & 31;
  if (wv >= nNodes) return;
  const float4* Hv = (const float4*)H;
  float sd = dinv[wv];
  float4 self = Hv[(size_t)wv * 32 + lane];
  float ax = self.x * sd, ay = self.y * sd, az = self.z * sd, aw = self.w * sd;
  int e0 = offs[wv], e1 = offs[wv + 1];
  for (int j = e0; j < e1; ++j) {
    int s = csrSrc[j];
    float w = dinv[s];
    float4 v = Hv[(size_t)s * 32 + lane];
    ax = fmaf(v.x, w, ax);
    ay = fmaf(v.y, w, ay);
    az = fmaf(v.z, w, az);
    aw = fmaf(v.w, w, aw);
  }
  float4 b = ((const float4*)bias)[lane];
  float4 o;
  o.x = fmaxf(fmaf(ax, sd, b.x), 0.0f);
  o.y = fmaxf(fmaf(ay, sd, b.y), 0.0f);
  o.z = fmaxf(fmaf(az, sd, b.z), 0.0f);
  o.w = fmaxf(fmaf(aw, sd, b.w), 0.0f);
  ((float4*)Y)[(size_t)wv * 32 + lane] = o;
}

// ---------------- segment-sum pool (batch is sorted) ----------------
__global__ __launch_bounds__(128)
void pool_kernel(const float* __restrict__ H, const int* __restrict__ batch,
                 float* __restrict__ pooled, float* __restrict__ gcnt, int nNodes) {
  const int col = threadIdx.x;
  const int base = blockIdx.x * 128;
  int curg = -1;
  float acc = 0.0f, count = 0.0f;
  for (int r = 0; r < 128; ++r) {
    int n = base + r;
    if (n >= nNodes) break;
    int g = batch[n];
    if (g != curg) {
      if (curg >= 0) {
        atomicAdd(&pooled[curg * DFEAT + col], acc);
        if (col == 0) atomicAdd(&gcnt[curg], count);
      }
      curg = g; acc = 0.0f; count = 0.0f;
    }
    acc += H[(size_t)n * DFEAT + col];
    count += 1.0f;
  }
  if (curg >= 0) {
    atomicAdd(&pooled[curg * DFEAT + col], acc);
    if (col == 0) atomicAdd(&gcnt[curg], count);
  }
}

// ---------------- final FC: out = relu( (pooled/cnt) @ Wfc + bfc ) ----------------
__global__ void fc_kernel(const float* __restrict__ pooled, const float* __restrict__ gcnt,
                          const float* __restrict__ Wfc, const float* __restrict__ bfc,
                          float* __restrict__ out, int G) {
  int idx = blockIdx.x * blockDim.x + threadIdx.x;
  if (idx >= G * DFEAT) return;
  int g = idx >> 7, c = idx & 127;
  float rinv = 1.0f / fmaxf(gcnt[g], 1.0f);
  float acc = 0.0f;
  for (int k = 0; k < DFEAT; ++k)
    acc = fmaf(pooled[g * DFEAT + k], Wfc[k * DFEAT + c], acc);
  out[idx] = fmaxf(fmaf(acc, rinv, bfc[c]), 0.0f);
}

// ---------------- launcher ----------------
extern "C" void kernel_launch(void* const* d_in, const int* in_sizes, int n_in,
                              void* d_out, int out_size, void* d_ws, size_t ws_size,
                              hipStream_t stream) {
  const float* x     = (const float*)d_in[0];
  const int*   ei    = (const int*)d_in[1];
  const int*   batch = (const int*)d_in[2];
  const float* W1    = (const float*)d_in[3];
  const float* b1    = (const float*)d_in[4];
  const float* W2    = (const float*)d_in[5];
  const float* b2    = (const float*)d_in[6];
  const float* Wfc   = (const float*)d_in[7];
  const float* bfc   = (const float*)d_in[8];

  const int N = in_sizes[0] / DFEAT;
  const int E = in_sizes[1] / 2;
  const int G = out_size / DFEAT;
  const int* src = ei;
  const int* dst = ei + E;

  char* ws = (char*)d_ws;
  size_t off = 0;
  auto alloc = [&](size_t bytes) -> void* {
    void* p = ws + off;
    off += (bytes + 255) & ~(size_t)255;
    return p;
  };
  float* bufA   = (float*)alloc((size_t)N * DFEAT * sizeof(float));
  float* bufB   = (float*)alloc((size_t)N * DFEAT * sizeof(float));
  float* dinv   = (float*)alloc((size_t)N * sizeof(float));
  int*   inDeg  = (int*)  alloc((size_t)N * sizeof(int));
  int*   offs   = (int*)  alloc(((size_t)N + 1) * sizeof(int));
  int*   cursor = (int*)  alloc(((size_t)N + 1) * sizeof(int));
  int*   csrSrc = (int*)  alloc((size_t)E * sizeof(int));
  float* pooled = (float*)alloc(((size_t)G * DFEAT + G) * sizeof(float));
  float* gcnt   = pooled + (size_t)G * DFEAT;
  (void)ws_size; (void)n_in;

  // graph structure (recomputed every call: deterministic work)
  zero_i32_kernel<<<(N + 255) / 256, 256, 0, stream>>>(inDeg, N);
  zero_f32_kernel<<<(G * DFEAT + G + 255) / 256, 256, 0, stream>>>(pooled, G * DFEAT + G);
  count_edges_kernel<<<(E + 255) / 256, 256, 0, stream>>>(dst, inDeg, E);
  scan_offsets_kernel<<<1, 1024, 0, stream>>>(inDeg, N, offs, cursor);
  dinv_kernel<<<(N + 255) / 256, 256, 0, stream>>>(inDeg, dinv, N);
  scatter_edges_kernel<<<(E + 255) / 256, 256, 0, stream>>>(src, dst, cursor, csrSrc, E);

  int nTiles = (N + 15) / 16;
  int gemmGrid = nTiles < 1024 ? nTiles : 1024;

  // layer 1
  gemm_xw_wmma_kernel<<<gemmGrid, 256, 0, stream>>>(x, W1, bufA, N);
  gcn_aggregate_kernel<<<(N + 7) / 8, 256, 0, stream>>>(bufA, offs, csrSrc, dinv, b1, bufB, N);
  // layer 2
  gemm_xw_wmma_kernel<<<gemmGrid, 256, 0, stream>>>(bufB, W2, bufA, N);
  gcn_aggregate_kernel<<<(N + 7) / 8, 256, 0, stream>>>(bufA, offs, csrSrc, dinv, b2, bufB, N);
  // pool + FC
  pool_kernel<<<(N + 127) / 128, 128, 0, stream>>>(bufB, batch, pooled, gcnt, N);
  fc_kernel<<<(G * DFEAT + 255) / 256, 256, 0, stream>>>(pooled, gcnt, Wfc, bfc, (float*)d_out, G);
}
